// MultiheadAttention_34411277975788
// MI455X (gfx1250) — compile-verified, matrix operand order checked by reading, not run on hardware
//
#include <hip/hip_runtime.h>
#include <stdint.h>

// ---------------------------------------------------------------------------
// Types
// ---------------------------------------------------------------------------
typedef __attribute__((ext_vector_type(16))) __bf16 v16bf;
typedef __attribute__((ext_vector_type(4)))  __bf16 v4bf;
typedef __attribute__((ext_vector_type(8)))  float  v8f;
typedef __attribute__((ext_vector_type(4)))  unsigned int u32x4;
typedef __attribute__((ext_vector_type(8)))  int    i32x8;
typedef __attribute__((ext_vector_type(4)))  int    i32x4;

#define S_LEN   2048
#define DMODEL  1024
#define NHEAD   16
#define HDIM    64
#define BATCH   4
#define MROWS   (BATCH * S_LEN)   // 8192

// ---------------------------------------------------------------------------
// WMMA wrapper: D(16x16 f32) = A(16x32 bf16) * B(32x16 bf16) + C
// ---------------------------------------------------------------------------
static __device__ __forceinline__ v8f wmma_bf16(v16bf a, v16bf b, v8f c) {
  return __builtin_amdgcn_wmma_f32_16x16x32_bf16(
      /*neg_a=*/false, a, /*neg_b=*/false, b,
      /*c_mod=*/(short)0, c, /*reuse_a=*/false, /*reuse_b=*/false);
}

// ---------------------------------------------------------------------------
// 16-lane-half reductions via ds_swizzle (xor stays inside each 16-lane half)
// offset encoding: [14:10]=xor, [9:5]=or, [4:0]=and
// ---------------------------------------------------------------------------
static __device__ __forceinline__ float half16_max(float x) {
  x = fmaxf(x, __int_as_float(__builtin_amdgcn_ds_swizzle(__float_as_int(x), 0x041F)));
  x = fmaxf(x, __int_as_float(__builtin_amdgcn_ds_swizzle(__float_as_int(x), 0x081F)));
  x = fmaxf(x, __int_as_float(__builtin_amdgcn_ds_swizzle(__float_as_int(x), 0x101F)));
  x = fmaxf(x, __int_as_float(__builtin_amdgcn_ds_swizzle(__float_as_int(x), 0x201F)));
  return x;
}
static __device__ __forceinline__ float half16_sum(float x) {
  x += __int_as_float(__builtin_amdgcn_ds_swizzle(__float_as_int(x), 0x041F));
  x += __int_as_float(__builtin_amdgcn_ds_swizzle(__float_as_int(x), 0x081F));
  x += __int_as_float(__builtin_amdgcn_ds_swizzle(__float_as_int(x), 0x101F));
  x += __int_as_float(__builtin_amdgcn_ds_swizzle(__float_as_int(x), 0x201F));
  return x;
}

// ---------------------------------------------------------------------------
// Tensor Data Mover: async 2D bf16 tile load Global -> LDS.
// D# packed per CDNA5 ISA ch.8 (group0: count/lds/global/type, group1: dims).
// 6-arg clang-23 builtin form: (u32x4 g0, i32x8 g1, i32x4 g2, i32x4 g3,
// i32x8 extra, i32 cpol). 2D tensor: groups 2/3 and the extra group are zero
// (tile_dim2=0 -> unused). Tile == tensor (no OOB), data_size=2B.
// ---------------------------------------------------------------------------
static __device__ __forceinline__ void tdm_load_2d_bf16(
    unsigned lds_byte_off, const void* gptr,
    unsigned width_elems, unsigned height_rows, unsigned row_stride_elems) {
  unsigned long long ga = (unsigned long long)(uintptr_t)gptr;
  u32x4 g0 = {0u, 0u, 0u, 0u};
  g0[0] = 1u;                                   // count=1, user descriptor
  g0[1] = lds_byte_off;                         // lds_addr
  g0[2] = (unsigned)ga;                         // global_addr[31:0]
  g0[3] = (unsigned)((ga >> 32) & 0x01FFFFFFu)  // global_addr[56:32]
          | 0x80000000u;                        // type=2 ("image")
  i32x8 g1 = {0, 0, 0, 0, 0, 0, 0, 0};
  g1[0] = (int)(1u << 16);                      // data_size=1 -> 2 bytes
  g1[1] = (int)((width_elems & 0xFFFFu) << 16);                 // tensor_dim0 lo
  g1[2] = (int)(((width_elems >> 16) & 0xFFFFu)
          | ((height_rows & 0xFFFFu) << 16));                   // dim0 hi | dim1 lo
  g1[3] = (int)(((height_rows >> 16) & 0xFFFFu)
          | ((width_elems & 0xFFFFu) << 16));                   // dim1 hi | tile_dim0
  g1[4] = (int)(height_rows & 0xFFFFu);                         // tile_dim1 (tile_dim2=0)
  g1[5] = (int)row_stride_elems;                                // tensor_dim0_stride lo32
  i32x4 gz4 = {0, 0, 0, 0};
  i32x8 gz8 = {0, 0, 0, 0, 0, 0, 0, 0};
  __builtin_amdgcn_tensor_load_to_lds(g0, g1, gz4, gz4, gz8, 0);
}

static __device__ __forceinline__ unsigned lds_off(const void* p) {
  return (unsigned)(unsigned long long)(uintptr_t)p;
}

// ---------------------------------------------------------------------------
// Kernel: f32 -> bf16 conversion, 4 elements / thread
// ---------------------------------------------------------------------------
__global__ __launch_bounds__(256) void k_cvt_bf16(
    const float* __restrict__ in, __bf16* __restrict__ out, int n4) {
  int i = blockIdx.x * blockDim.x + threadIdx.x;
  if (i < n4) {
    float4 f = ((const float4*)in)[i];
    v4bf o;
    o[0] = (__bf16)f.x; o[1] = (__bf16)f.y;
    o[2] = (__bf16)f.z; o[3] = (__bf16)f.w;
    ((v4bf*)out)[i] = o;
  }
}

// ---------------------------------------------------------------------------
// Kernel: W[k][n] (f32, 1024x1024) -> WT[n][k] (bf16) via LDS tile transpose
// ---------------------------------------------------------------------------
__global__ __launch_bounds__(256) void k_transpose_cvt(
    const float* __restrict__ W, __bf16* __restrict__ WT) {
  __shared__ float tile[32][33];
  int bx = blockIdx.x * 32, by = blockIdx.y * 32;
  int tx = threadIdx.x & 31, ty = threadIdx.x >> 5;
  #pragma unroll
  for (int i = 0; i < 32; i += 8)
    tile[ty + i][tx] = W[(size_t)(by + ty + i) * DMODEL + bx + tx];
  __syncthreads();
  #pragma unroll
  for (int i = 0; i < 32; i += 8)
    WT[(size_t)(bx + ty + i) * DMODEL + by + tx] = (__bf16)tile[tx][ty + i];
}

// ---------------------------------------------------------------------------
// Kernel: C = A[M x 1024] * BT^T + bias, WMMA bf16->f32.
// Block = 256 thr = 8 waves (4 along M, 2 along N); block tile 128x128;
// wave tile 32x64; K-step 32. A and BT both row-major with contiguous
// per-lane 32B fragment loads (BT holds B transposed).
// mode 0: scatter bf16 to [B,H,S,64] (Q/K)
// mode 1: scatter bf16 to [B,H,64,S] (V transposed)
// mode 2: f32 row-major [M x 1024]   (final output)
// ---------------------------------------------------------------------------
__global__ __launch_bounds__(256) void k_gemm_wmma(
    const __bf16* __restrict__ A, const __bf16* __restrict__ BT,
    const float* __restrict__ bias, void* __restrict__ out, int mode) {
  const int K = DMODEL;
  int lane = threadIdx.x & 31;
  int wave = threadIdx.x >> 5;
  int wm = wave >> 1, wn = wave & 1;
  int m0 = blockIdx.x * 128 + wm * 32;
  int n0 = blockIdx.y * 128 + wn * 64;
  int lr = lane & 15;            // fragment row / column within 16
  int lk = (lane >> 4) * 16;     // K sub-range select (0 or 16)
  int lh = lane >> 4;            // lane half

  v8f acc[2][4] = {};
  for (int k0 = 0; k0 < K; k0 += 32) {
    v16bf a[2], b[4];
    #pragma unroll
    for (int i = 0; i < 2; ++i)
      a[i] = *(const v16bf*)(A + (size_t)(m0 + i * 16 + lr) * K + k0 + lk);
    #pragma unroll
    for (int j = 0; j < 4; ++j)
      b[j] = *(const v16bf*)(BT + (size_t)(n0 + j * 16 + lr) * K + k0 + lk);
    // prefetch next K-slab into L2 (speculative; OOB drops silently)
    __builtin_prefetch(A + (size_t)(m0 + lr) * K + k0 + 32 + lk, 0, 1);
    __builtin_prefetch(BT + (size_t)(n0 + lr) * K + k0 + 32 + lk, 0, 1);
    #pragma unroll
    for (int i = 0; i < 2; ++i)
      #pragma unroll
      for (int j = 0; j < 4; ++j)
        acc[i][j] = wmma_bf16(a[i], b[j], acc[i][j]);
  }

  // Epilogue. D layout: VGPR r -> M = r (lanes 0-15) / r+8 (lanes 16-31), N = lane%16.
  #pragma unroll
  for (int i = 0; i < 2; ++i)
    #pragma unroll
    for (int j = 0; j < 4; ++j)
      #pragma unroll
      for (int r = 0; r < 8; ++r) {
        int row = m0 + i * 16 + r + lh * 8;
        int col = n0 + j * 16 + lr;
        float vv = acc[i][j][r] + bias[col];
        if (mode == 2) {
          ((float*)out)[(size_t)row * DMODEL + col] = vv;
        } else {
          int bb = row >> 11, ss = row & (S_LEN - 1);
          int hh = col >> 6,  dd = col & (HDIM - 1);
          if (mode == 0)
            ((__bf16*)out)[(((size_t)bb * NHEAD + hh) * S_LEN + ss) * HDIM + dd] = (__bf16)vv;
          else
            ((__bf16*)out)[(((size_t)bb * NHEAD + hh) * HDIM + dd) * S_LEN + ss] = (__bf16)vv;
        }
      }
}

// ---------------------------------------------------------------------------
// Flash attention. Grid (S/64, B*H); block = 128 thr = 4 waves.
// Each wave owns 16 query rows. K/V 64-key tiles stream through DOUBLE-
// BUFFERED LDS via the Tensor Data Mover: wave 0 issues tile i+1's two TDM
// loads, then waits tensorcnt<=2 -- TDM ops complete in order per wave, so
// tile i is guaranteed resident while tile i+1's DMA overlaps the 16 WMMAs
// + softmax of tile i.
//   Q    : [B,H,S,64]  bf16
//   Kmat : [B,H,S,64]  bf16
//   VT   : [B,H,64,S]  bf16 (V transposed so PV B-frags are contiguous)
//   heads: [B,S,H,64]  bf16 (== [8192,1024] rows for the output GEMM)
// LDS: 2*8KB (K) + 2*8KB (V) + 8KB (P) = 40 KB / workgroup.
// ---------------------------------------------------------------------------
__global__ __launch_bounds__(128) void k_flash_attn(
    const __bf16* __restrict__ Q, const __bf16* __restrict__ Kmat,
    const __bf16* __restrict__ VT, __bf16* __restrict__ heads) {
  __shared__ __bf16 sK[2][64 * 64];     // [buf][key][d]
  __shared__ __bf16 sV[2][64 * 64];     // [buf][d][key]   (VT tile)
  __shared__ __bf16 sP[4][16 * 64];     // per-wave P staging [row][key]

  int lane = threadIdx.x & 31;
  int wave = threadIdx.x >> 5;
  int bh = blockIdx.y;                          // 0..63 == b*16+h
  int q0 = blockIdx.x * 64 + wave * 16;         // first query row of this wave
  int lr = lane & 15;
  int lk = (lane >> 4) * 16;
  int lh = lane >> 4;

  const __bf16* Kbase = Kmat + (size_t)bh * S_LEN * HDIM;   // [key][d]
  const __bf16* Vbase = VT + (size_t)bh * HDIM * S_LEN;     // [d][key]

  // Q strip 16x64: two A-frags, loaded once (contiguous 32B per lane).
  const __bf16* Qbase = Q + ((size_t)bh * S_LEN + q0) * HDIM;
  v16bf aq[2];
  aq[0] = *(const v16bf*)(Qbase + (size_t)lr * HDIM + lk);
  aq[1] = *(const v16bf*)(Qbase + (size_t)lr * HDIM + 32 + lk);

  v8f acc_o[4] = {};
  float m_i[8], l_i[8];
  #pragma unroll
  for (int r = 0; r < 8; ++r) { m_i[r] = -1e30f; l_i[r] = 0.0f; }
  const float scale = 0.125f;   // 1/sqrt(64)

  const int NT = S_LEN / 64;    // 32 key tiles
  // Prologue: kick off tile 0 into buffer 0.
  if (wave == 0) {
    tdm_load_2d_bf16(lds_off(sK[0]), Kbase, 64, 64, HDIM);
    tdm_load_2d_bf16(lds_off(sV[0]), Vbase, 64, 64, S_LEN);
  }

  for (int it = 0; it < NT; ++it) {
    int cur = it & 1, nxt = cur ^ 1;
    __syncthreads();   // buf[nxt] (tile it-1) fully consumed by all waves
    if (wave == 0) {
      if (it + 1 < NT) {
        tdm_load_2d_bf16(lds_off(sK[nxt]),
                         Kbase + (size_t)(it + 1) * 64 * HDIM, 64, 64, HDIM);
        tdm_load_2d_bf16(lds_off(sV[nxt]),
                         Vbase + (it + 1) * 64, 64, 64, S_LEN);
        __builtin_amdgcn_s_wait_tensorcnt(2);   // tile `it` done, tile it+1 in flight
      } else {
        __builtin_amdgcn_s_wait_tensorcnt(0);   // last tile: drain
      }
    }
    __syncthreads();   // tile `it` visible to all waves

    // Scores: S16x64 = Q(16x64) . K_tile^T ; B-frag = rows of K tile.
    v8f sc[4] = {};
    #pragma unroll
    for (int c = 0; c < 4; ++c)
      #pragma unroll
      for (int kq = 0; kq < 2; ++kq) {
        v16bf bk = *(const v16bf*)(&sK[cur][(c * 16 + lr) * 64 + kq * 32 + lk]);
        sc[c] = wmma_bf16(aq[kq], bk, sc[c]);
      }

    // Online softmax per row (row = r + 8*lane_half, cols across 16 lanes).
    #pragma unroll
    for (int r = 0; r < 8; ++r) {
      float rm = -1e30f;
      #pragma unroll
      for (int c = 0; c < 4; ++c) rm = fmaxf(rm, sc[c][r]);
      rm = half16_max(rm) * scale;
      float mnew = fmaxf(m_i[r], rm);
      float alpha = __expf(m_i[r] - mnew);
      float rs = 0.0f;
      #pragma unroll
      for (int c = 0; c < 4; ++c) {
        float p = __expf(sc[c][r] * scale - mnew);
        sc[c][r] = p;
        rs += p;
      }
      rs = half16_sum(rs);
      l_i[r] = l_i[r] * alpha + rs;
      m_i[r] = mnew;
      #pragma unroll
      for (int c2 = 0; c2 < 4; ++c2) acc_o[c2][r] *= alpha;
      // stage P (D-layout) into LDS in row-major form
      int prow = r + lh * 8;
      #pragma unroll
      for (int c = 0; c < 4; ++c)
        sP[wave][prow * 64 + c * 16 + lr] = (__bf16)sc[c][r];
    }

    // Reload P as A-frags (wave-private LDS; DS ops in-order within wave).
    v16bf ap[2];
    ap[0] = *(const v16bf*)(&sP[wave][lr * 64 + lk]);
    ap[1] = *(const v16bf*)(&sP[wave][lr * 64 + 32 + lk]);

    // O += P(16x64keys) . V(64keys x 64d) ; B-frag = rows of sV (= V^T tile).
    #pragma unroll
    for (int c2 = 0; c2 < 4; ++c2)
      #pragma unroll
      for (int k2 = 0; k2 < 2; ++k2) {
        v16bf bv = *(const v16bf*)(&sV[cur][(c2 * 16 + lr) * 64 + k2 * 32 + lk]);
        acc_o[c2] = wmma_bf16(ap[k2], bv, acc_o[c2]);
      }
  }

  // Normalize and write heads[b, s, h, d] (bf16).
  int bb = bh >> 4, hh = bh & 15;
  #pragma unroll
  for (int r = 0; r < 8; ++r) {
    float inv = 1.0f / l_i[r];
    int ss = q0 + r + lh * 8;
    #pragma unroll
    for (int c2 = 0; c2 < 4; ++c2) {
      int dd = c2 * 16 + lr;
      heads[(((size_t)bb * S_LEN + ss) * NHEAD + hh) * HDIM + dd] =
          (__bf16)(acc_o[c2][r] * inv);
    }
  }
}

// ---------------------------------------------------------------------------
// Host orchestration
// ---------------------------------------------------------------------------
extern "C" void kernel_launch(void* const* d_in, const int* in_sizes, int n_in,
                              void* d_out, int out_size, void* d_ws, size_t ws_size,
                              hipStream_t stream) {
  (void)in_sizes; (void)n_in; (void)out_size; (void)ws_size;
  const float* x  = (const float*)d_in[0];
  const float* Wq = (const float*)d_in[1];
  const float* bq = (const float*)d_in[2];
  const float* Wk = (const float*)d_in[3];
  const float* bk = (const float*)d_in[4];
  const float* Wv = (const float*)d_in[5];
  const float* bv = (const float*)d_in[6];
  const float* Wo = (const float*)d_in[7];
  const float* bo = (const float*)d_in[8];

  const size_t ACT = (size_t)MROWS * DMODEL * sizeof(__bf16);   // 16 MB
  const size_t WSZ = (size_t)DMODEL * DMODEL * sizeof(__bf16);  //  2 MB
  char* p = (char*)d_ws;
  __bf16* xbf   = (__bf16*)p; p += ACT;
  __bf16* WqT   = (__bf16*)p; p += WSZ;
  __bf16* WkT   = (__bf16*)p; p += WSZ;
  __bf16* WvT   = (__bf16*)p; p += WSZ;
  __bf16* WoT   = (__bf16*)p; p += WSZ;
  __bf16* Qb    = (__bf16*)p; p += ACT;   // [B,H,S,64]
  __bf16* Kb    = (__bf16*)p; p += ACT;   // [B,H,S,64]
  __bf16* VTb   = (__bf16*)p; p += ACT;   // [B,H,64,S]
  __bf16* hds   = (__bf16*)p; p += ACT;   // [B,S,H,64]

  // 1) precision conversion / weight transposition
  int n4 = MROWS * DMODEL / 4;
  k_cvt_bf16<<<(n4 + 255) / 256, 256, 0, stream>>>(x, xbf, n4);
  dim3 tg(32, 32);
  k_transpose_cvt<<<tg, 256, 0, stream>>>(Wq, WqT);
  k_transpose_cvt<<<tg, 256, 0, stream>>>(Wk, WkT);
  k_transpose_cvt<<<tg, 256, 0, stream>>>(Wv, WvT);
  k_transpose_cvt<<<tg, 256, 0, stream>>>(Wo, WoT);

  // 2) QKV projections (WMMA)
  dim3 gg(MROWS / 128, DMODEL / 128);   // (64, 8)
  k_gemm_wmma<<<gg, 256, 0, stream>>>(xbf, WqT, bq, (void*)Qb, 0);
  k_gemm_wmma<<<gg, 256, 0, stream>>>(xbf, WkT, bk, (void*)Kb, 0);
  k_gemm_wmma<<<gg, 256, 0, stream>>>(xbf, WvT, bv, (void*)VTb, 1);

  // 3) flash attention (WMMA + double-buffered TDM pipeline)
  dim3 ga(S_LEN / 64, BATCH * NHEAD);   // (32, 64)
  k_flash_attn<<<ga, 128, 0, stream>>>(Qb, Kb, VTb, hds);

  // 4) output projection, f32 result
  k_gemm_wmma<<<gg, 256, 0, stream>>>(hds, WoT, bo, d_out, 2);
}
